// NBCModel_32366873542800
// MI455X (gfx1250) — matrix operand order repaired and checked by reading, not compiled, not run on hardware
//
#include <hip/hip_runtime.h>
#include <hip/hip_bf16.h>

// ---------------------------------------------------------------------------
// NBC walk-count model on MI455X (gfx1250, wave32, WMMA).
//
// diag(B^k) for k=1..8 via half-power trick: only B^2, B^3, B^4 are formed as
// full matrices (3 f16 WMMA GEMMs, f32 accumulate -> exact for small integer
// walk counts). Diagonals come from coalesced row-dot-row products against
// stored transposes. Then segment-sum + tiny MLP.
// ---------------------------------------------------------------------------

typedef __attribute__((ext_vector_type(16))) _Float16 v16h;
typedef __attribute__((ext_vector_type(8)))  float    v8f;

#define E_DIM 2048
#define L_DIM 8
#define G_DIM 128
#define HID_DIM 128

// ---------------------------------------------------------------------------
// Build B and B^T as f16. B[i,j] = (dst[j]==src[i] && src[j]!=dst[i]).
// ---------------------------------------------------------------------------
__global__ __launch_bounds__(256) void build_B_kernel(
    const int* __restrict__ src, const int* __restrict__ dst,
    _Float16* __restrict__ B, _Float16* __restrict__ BT) {
  const int idx = blockIdx.x * blockDim.x + threadIdx.x;   // over E*E
  const int i = idx >> 11;          // row (edge i)
  const int j = idx & (E_DIM - 1);  // col (edge j)
  const int si = src[i], di = dst[i];
  const int sj = src[j], dj = dst[j];
  const _Float16 v = (_Float16)(((dj == si) && (sj != di)) ? 1.0f : 0.0f);
  B[(size_t)i * E_DIM + j] = v;
  BT[(size_t)j * E_DIM + i] = v;
}

// ---------------------------------------------------------------------------
// C = A * Bmat, where the second operand is supplied as BT (row-major
// transpose of Bmat). All matrices E_DIM x E_DIM, f16 storage, f32 accumulate.
//
// 128 threads = 4 waves; block tile 128x128; wave tile 64x64
// => 16 x v_wmma_f32_16x16x32_f16 per K step, K loop = 2048/32 = 64.
//
// Fragment layouts follow CDNA5 ISA 7.12.2 (16-bit, 16x32 A / 32x16 B):
//   A: lane l (0-15) = row m0+16t+l ; halves [0..7]=K+0..7, [8..15]=K+16..23
//      lane l+16     = same row     ; halves [0..7]=K+8..15, [8..15]=K+24..31
//   B: lane l (0-15) = col n0+16t+l ; halves [0..7]=K+0..7, [8..15]=K+16..23 (rows of BT)
//      lane l+16     = same col     ; halves [0..7]=K+8..15, [8..15]=K+24..31
// All per-lane loads are two contiguous 16B chunks -> global_load_b128.
// ---------------------------------------------------------------------------
__global__ __launch_bounds__(128) void gemm_f16_wmma(
    const _Float16* __restrict__ A, const _Float16* __restrict__ BT,
    _Float16* __restrict__ C) {
  const int tid  = threadIdx.x;
  const int wave = tid >> 5;
  const int lane = tid & 31;
  const int l    = lane & 15;
  const int hi   = lane >> 4;

  const int m0 = blockIdx.y * 128 + (wave >> 1) * 64;
  const int n0 = blockIdx.x * 128 + (wave & 1) * 64;

  const v8f vzero = {0.f, 0.f, 0.f, 0.f, 0.f, 0.f, 0.f, 0.f};
  v8f acc[4][4];
#pragma unroll
  for (int i = 0; i < 4; ++i)
#pragma unroll
    for (int j = 0; j < 4; ++j) acc[i][j] = vzero;

  for (int k0 = 0; k0 < E_DIM; k0 += 32) {
    // prefetch next K tile (emits global_prefetch_b8)
    if (k0 + 32 < E_DIM) {
      __builtin_prefetch(A  + (size_t)(m0 + l) * E_DIM + k0 + 32, 0, 1);
      __builtin_prefetch(BT + (size_t)(n0 + l) * E_DIM + k0 + 32, 0, 1);
    }

    v16h a[4], b[4];
#pragma unroll
    for (int t = 0; t < 4; ++t) {
      const int r = m0 + 16 * t + l;
      const _Float16* pa = A + (size_t)r * E_DIM + k0 + hi * 8;
      union { uint4 u[2]; v16h v; } ua;
      ua.u[0] = *(const uint4*)(pa);        // K chunk 0 (8 halves)
      ua.u[1] = *(const uint4*)(pa + 16);   // K chunk +16
      a[t] = ua.v;
    }
#pragma unroll
    for (int t = 0; t < 4; ++t) {
      const int c = n0 + 16 * t + l;
      const _Float16* pb = BT + (size_t)c * E_DIM + k0 + hi * 8;
      union { uint4 u[2]; v16h v; } ub;
      ub.u[0] = *(const uint4*)(pb);        // K chunk 0 (8 halves)
      ub.u[1] = *(const uint4*)(pb + 16);   // K chunk +16
      b[t] = ub.v;
    }

#pragma unroll
    for (int i = 0; i < 4; ++i)
#pragma unroll
      for (int j = 0; j < 4; ++j)
        acc[i][j] = __builtin_amdgcn_wmma_f32_16x16x32_f16(
            false, a[i], false, b[j], (short)0, acc[i][j], false, false);
  }

  // C/D layout: VGPR v -> row m0+16i + (hi?8:0) + v ; col n0+16j + l
#pragma unroll
  for (int i = 0; i < 4; ++i) {
#pragma unroll
    for (int j = 0; j < 4; ++j) {
      const int col = n0 + 16 * j + l;
#pragma unroll
      for (int v = 0; v < 8; ++v) {
        const int row = m0 + 16 * i + hi * 8 + v;
        C[(size_t)row * E_DIM + col] = (_Float16)acc[i][j][v];
      }
    }
  }
}

// ---------------------------------------------------------------------------
// 64x64 LDS-tiled f16 transpose.
// ---------------------------------------------------------------------------
__global__ __launch_bounds__(256) void transpose_f16(
    const _Float16* __restrict__ in, _Float16* __restrict__ out) {
  __shared__ _Float16 tile[64][65];
  const int bx = blockIdx.x * 64, by = blockIdx.y * 64;
  for (int i = threadIdx.y; i < 64; i += 8)
    for (int j = threadIdx.x; j < 64; j += 32)
      tile[i][j] = in[(size_t)(by + i) * E_DIM + bx + j];
  __syncthreads();
  for (int i = threadIdx.y; i < 64; i += 8)
    for (int j = threadIdx.x; j < 64; j += 32)
      out[(size_t)(bx + i) * E_DIM + by + j] = tile[j][i];
}

// ---------------------------------------------------------------------------
// All 8 diagonals. diag(B^(p+q))[i] = dot(row_i(B^p), row_i((B^q)^T)).
// All reads are coalesced rows. One block per edge-row i.
// diags layout: [L][E] (matches reference's diags).
// ---------------------------------------------------------------------------
__global__ __launch_bounds__(256) void diag_all_kernel(
    const _Float16* __restrict__ B,  const _Float16* __restrict__ BT,
    const _Float16* __restrict__ B2, const _Float16* __restrict__ B2T,
    const _Float16* __restrict__ B3, const _Float16* __restrict__ B3T,
    const _Float16* __restrict__ B4, const _Float16* __restrict__ B4T,
    float* __restrict__ diags) {
  const int i = blockIdx.x;
  const int tid = threadIdx.x;
  const size_t base = (size_t)i * E_DIM;

  float s[7] = {0.f, 0.f, 0.f, 0.f, 0.f, 0.f, 0.f};
  for (int j = tid; j < E_DIM; j += 256) {
    const float b   = (float)B [base + j];
    const float bt  = (float)BT[base + j];
    const float b2  = (float)B2[base + j];
    const float b2t = (float)B2T[base + j];
    const float b3  = (float)B3[base + j];
    const float b3t = (float)B3T[base + j];
    const float b4  = (float)B4[base + j];
    const float b4t = (float)B4T[base + j];
    s[0] += b  * bt;    // diag(B^2)
    s[1] += b2 * bt;    // diag(B^3)
    s[2] += b2 * b2t;   // diag(B^4)
    s[3] += b3 * b2t;   // diag(B^5)
    s[4] += b3 * b3t;   // diag(B^6)
    s[5] += b4 * b3t;   // diag(B^7)
    s[6] += b4 * b4t;   // diag(B^8)
  }

  __shared__ float red[7][256];
  for (int k = 0; k < 7; ++k) red[k][tid] = s[k];
  __syncthreads();
  for (int stride = 128; stride > 0; stride >>= 1) {
    if (tid < stride)
      for (int k = 0; k < 7; ++k) red[k][tid] += red[k][tid + stride];
    __syncthreads();
  }
  if (tid == 0) {
    diags[0 * E_DIM + i] = (float)B[base + i];          // diag(B^1)
    for (int k = 0; k < 7; ++k) diags[(k + 1) * E_DIM + i] = red[k][0];
  }
}

// ---------------------------------------------------------------------------
// Per-graph segment-sum of diagonals + MLP (relu(sig@W1+b1)@W2+b2).
// One block (128 threads) per graph. Deterministic (no atomics).
// ---------------------------------------------------------------------------
__global__ __launch_bounds__(128) void seg_mlp_kernel(
    const float* __restrict__ diags, const int* __restrict__ edge_graph,
    const float* __restrict__ W1, const float* __restrict__ b1,
    const float* __restrict__ W2, const float* __restrict__ b2,
    float* __restrict__ out) {
  const int g = blockIdx.x;
  const int tid = threadIdx.x;

  float s[L_DIM] = {0.f, 0.f, 0.f, 0.f, 0.f, 0.f, 0.f, 0.f};
  for (int e = tid; e < E_DIM; e += 128) {
    if (edge_graph[e] == g) {
#pragma unroll
      for (int l = 0; l < L_DIM; ++l) s[l] += diags[l * E_DIM + e];
    }
  }

  __shared__ float red[L_DIM][128];
  for (int l = 0; l < L_DIM; ++l) red[l][tid] = s[l];
  __syncthreads();
  for (int stride = 64; stride > 0; stride >>= 1) {
    if (tid < stride)
      for (int l = 0; l < L_DIM; ++l) red[l][tid] += red[l][tid + stride];
    __syncthreads();
  }

  __shared__ float sig[L_DIM];
  if (tid < L_DIM) sig[tid] = red[tid][0];
  __syncthreads();

  // h[tid] = relu(sig @ W1 + b1)[tid], tid in [0,HID)
  float h = b1[tid];
#pragma unroll
  for (int l = 0; l < L_DIM; ++l) h += sig[l] * W1[l * HID_DIM + tid];
  h = h > 0.f ? h : 0.f;

  __shared__ float red2[128];
  red2[tid] = h * W2[tid];
  __syncthreads();
  for (int stride = 64; stride > 0; stride >>= 1) {
    if (tid < stride) red2[tid] += red2[tid + stride];
    __syncthreads();
  }
  if (tid == 0) out[g] = red2[0] + b2[0];
}

// ---------------------------------------------------------------------------
extern "C" void kernel_launch(void* const* d_in, const int* in_sizes, int n_in,
                              void* d_out, int out_size, void* d_ws, size_t ws_size,
                              hipStream_t stream) {
  (void)in_sizes; (void)n_in; (void)out_size; (void)ws_size;

  const int* edge_index = (const int*)d_in[0];   // [2, E]
  const int* src = edge_index;
  const int* dst = edge_index + E_DIM;
  const int* edge_graph = (const int*)d_in[1];   // [E]
  const float* W1 = (const float*)d_in[2];       // [L, HID]
  const float* b1 = (const float*)d_in[3];       // [HID]
  const float* W2 = (const float*)d_in[4];       // [HID, 1]
  const float* b2 = (const float*)d_in[5];       // [1]
  float* out = (float*)d_out;                    // [G, 1]

  const size_t EE = (size_t)E_DIM * E_DIM;
  _Float16* ws16 = (_Float16*)d_ws;
  _Float16* B   = ws16 + 0 * EE;
  _Float16* BT  = ws16 + 1 * EE;
  _Float16* B2  = ws16 + 2 * EE;
  _Float16* B2T = ws16 + 3 * EE;
  _Float16* B3  = ws16 + 4 * EE;
  _Float16* B3T = ws16 + 5 * EE;
  _Float16* B4  = ws16 + 6 * EE;
  _Float16* B4T = ws16 + 7 * EE;
  float* diags = (float*)(ws16 + 8 * EE);        // [L, E]

  // 1. Build B, B^T
  build_B_kernel<<<(int)(EE / 256), 256, 0, stream>>>(src, dst, B, BT);

  const dim3 ggrid(E_DIM / 128, E_DIM / 128);    // 16 x 16
  const dim3 tgrid(E_DIM / 64, E_DIM / 64);      // 32 x 32
  const dim3 tblk(32, 8);

  // 2. B2 = B * B    (operand supplied transposed: BT)
  gemm_f16_wmma<<<ggrid, 128, 0, stream>>>(B, BT, B2);
  transpose_f16<<<tgrid, tblk, 0, stream>>>(B2, B2T);

  // 3. B3 = B2 * B
  gemm_f16_wmma<<<ggrid, 128, 0, stream>>>(B2, BT, B3);
  transpose_f16<<<tgrid, tblk, 0, stream>>>(B3, B3T);

  // 4. B4 = B2 * B2
  gemm_f16_wmma<<<ggrid, 128, 0, stream>>>(B2, B2T, B4);
  transpose_f16<<<tgrid, tblk, 0, stream>>>(B4, B4T);

  // 5. All 8 diagonals
  diag_all_kernel<<<E_DIM, 256, 0, stream>>>(B, BT, B2, B2T, B3, B3T, B4, B4T,
                                             diags);

  // 6. Segment sum + MLP
  seg_mlp_kernel<<<G_DIM, 128, 0, stream>>>(diags, edge_graph, W1, b1, W2, b2,
                                            out);
}
